// VectorQuantizer_54013508714591
// MI455X (gfx1250) — matrix operand order repaired and checked by reading, not compiled, not run on hardware
//
#include <hip/hip_runtime.h>

// ---------------- types ----------------
typedef __attribute__((ext_vector_type(16))) __bf16 v16bf;
typedef __attribute__((ext_vector_type(8)))  __bf16 v8bf;
typedef __attribute__((ext_vector_type(4)))  __bf16 v4bf;
typedef __attribute__((ext_vector_type(8)))  float  v8f;
typedef __attribute__((ext_vector_type(4)))  int    v4i;

#define D        512      // embedding dim (K)
#define NE       8192     // number of codes (N)
#define MROWS    16384    // number of queries (M) = 8*2048
#define MT       128      // query rows per block
#define NCHUNK   128      // codes per N-chunk
#define KSLICE   64       // K slice staged per B load
#define SA       520      // A LDS stride in bf16 (260 dwords == 4 mod 64 -> conflict-free frags)
#define SB       72       // B LDS stride in bf16 (36 dwords -> 16 distinct banks over 16 lanes)

// CDNA5 async global->LDS path (ASYNCcnt-tracked), with safe fallback.
#if defined(__AMDGCN__) && __has_builtin(__builtin_amdgcn_global_load_async_to_lds_b128)
#define USE_ASYNC 1
typedef __attribute__((address_space(1))) v4i* gv4i_p;   // global int4*
typedef __attribute__((address_space(3))) v4i* lv4i_p;   // LDS    int4*
#else
#define USE_ASYNC 0
#endif

__device__ __forceinline__ v8f wmma_bf16(v16bf a, v16bf b, v8f c) {
  // D = A*B + C, f32 accumulate, 16x16x32 bf16
  return __builtin_amdgcn_wmma_f32_16x16x32_bf16(false, a, false, b, (short)0, c, false, false);
}

// Load a 16-element bf16 fragment as two 16B LDS reads (ds_load_b128 x2).
// gap = 8 for both A and B fragments here (16 contiguous K per lane).
__device__ __forceinline__ v16bf ld_frag(const __bf16* p, int gap) {
  v8bf a = *(const v8bf*)p;
  v8bf b = *(const v8bf*)(p + gap);
  return __builtin_shufflevector(a, b, 0,1,2,3,4,5,6,7,8,9,10,11,12,13,14,15);
}

__device__ __forceinline__ void split_store(__bf16* hi, __bf16* lo, int off, float x) {
  __bf16 h = (__bf16)x;
  hi[off] = h;
  lo[off] = (__bf16)(x - (float)h);
}

// 16-byte global->LDS copy: async DMA on CDNA5, VGPR bounce otherwise.
__device__ __forceinline__ void cp16_g2l(const __bf16* g, __bf16* l) {
#if USE_ASYNC
  __builtin_amdgcn_global_load_async_to_lds_b128((gv4i_p)g, (lv4i_p)l, 0, 0);
#else
  *(v8bf*)l = *(const v8bf*)g;
#endif
}

__device__ __forceinline__ void wait_async() {
#if USE_ASYNC
#if __has_builtin(__builtin_amdgcn_s_wait_asynccnt)
  __builtin_amdgcn_s_wait_asynccnt(0);
#else
  asm volatile("s_wait_asynccnt 0x0" ::: "memory");
#endif
#endif
}

// ---------------- kernel 0a: codebook f32 -> planar bf16 hi/lo (run once) ----------------
__global__ __launch_bounds__(256) void vq_cvt_kernel(const float* __restrict__ cb,
                                                     __bf16* __restrict__ cbhi,
                                                     __bf16* __restrict__ cblo) {
  size_t g = ((size_t)blockIdx.x * 256 + threadIdx.x) * 4;
  float4 v = *(const float4*)(cb + g);
  v4bf h, l;
  __bf16 t;
  t = (__bf16)v.x; h[0] = t; l[0] = (__bf16)(v.x - (float)t);
  t = (__bf16)v.y; h[1] = t; l[1] = (__bf16)(v.y - (float)t);
  t = (__bf16)v.z; h[2] = t; l[2] = (__bf16)(v.z - (float)t);
  t = (__bf16)v.w; h[3] = t; l[3] = (__bf16)(v.w - (float)t);
  *(v4bf*)(cbhi + g) = h;
  *(v4bf*)(cblo + g) = l;
}

// ---------------- kernel 0b: codebook squared norms (exact f32) ----------------
__global__ __launch_bounds__(256) void vq_enorm_kernel(const float* __restrict__ cb,
                                                       float* __restrict__ enorm) {
  int n = blockIdx.x * 256 + threadIdx.x;   // 8192 threads total
  const float* r = cb + (size_t)n * D;
  float s = 0.f;
  #pragma unroll 4
  for (int k = 0; k < D; k += 4) {
    float4 v = *(const float4*)(r + k);
    s += v.x * v.x + v.y * v.y + v.z * v.z + v.w * v.w;
  }
  enorm[n] = s;
}

// ---------------- kernel 1: fused distance GEMM + argmin ----------------
// grid = MROWS/MT = 128 blocks, 256 threads (8 waves), dynamic LDS ~303 KB (of 320 KB/WGP).
// Wave (wm, wn): wm = wave>>1 owns rows wm*32..+31, wn = wave&1 owns 64-wide N half of chunk.
__global__ __launch_bounds__(256) void vq_argmin_kernel(const float* __restrict__ z,
                                                        const __bf16* __restrict__ cbhi,
                                                        const __bf16* __restrict__ cblo,
                                                        const float* __restrict__ enorm,
                                                        int* __restrict__ idx_out,
                                                        float* __restrict__ idxf_out) {
  extern __shared__ char smem[];
  __bf16* sAhi = (__bf16*)smem;                 // MT * SA
  __bf16* sAlo = sAhi + MT * SA;                // MT * SA
  __bf16* sBhi = sAlo + MT * SA;                // NCHUNK * SB
  __bf16* sBlo = sBhi + NCHUNK * SB;            // NCHUNK * SB
  float2* sRed = (float2*)sBhi;                 // reused after GEMM: 128*2 float2 (2 KB)

  const int tid  = threadIdx.x;
  const int lane = tid & 31;
  const int wave = tid >> 5;
  const int wm   = wave >> 1;        // 0..3  M group
  const int wn   = wave & 1;         // 0..1  N half
  const int half = lane >> 4;        // 0..1
  const int l16  = lane & 15;
  const int blk  = blockIdx.x;

  // ---- stage A: 128 rows x 512 K, f32 -> bf16 hi/lo in LDS (once per block) ----
  const float* zb = z + (size_t)blk * MT * D;
  #pragma unroll 4
  for (int it = 0; it < (MT * D / 4) / 256; ++it) {   // 64 iters of float4
    int f = (tid + it * 256) * 4;
    int r = f >> 9, k = f & (D - 1);
    float4 v = *(const float4*)(zb + f);
    int off = r * SA + k;
    split_store(sAhi, sAlo, off + 0, v.x);
    split_store(sAhi, sAlo, off + 1, v.y);
    split_store(sAhi, sAlo, off + 2, v.z);
    split_store(sAhi, sAlo, off + 3, v.w);
  }

  float bestv[16];
  int   besti[16];
  #pragma unroll
  for (int i = 0; i < 16; ++i) { bestv[i] = 3.4e38f; besti[i] = 0; }

  const int am0 = wm * 32 + l16;      // A tile0 row (tile1 = +16)
  const int bnb = wn * 64 + l16;      // B base column within chunk

  for (int c = 0; c < NE / NCHUNK; ++c) {            // 64 N-chunks
    v8f acc[8];
    #pragma unroll
    for (int i = 0; i < 8; ++i) acc[i] = (v8f)(0.0f);

    for (int ks = 0; ks < D / KSLICE; ++ks) {        // 8 K-slices
      __syncthreads();   // previous compute done before overwriting B (also fences A stage)
      // ---- stage B slice: pure 16B copies of pre-converted bf16 (no VALU conversion) ----
      const __bf16* gh = cbhi + (size_t)(c * NCHUNK) * D + ks * KSLICE;
      const __bf16* gl = cblo + (size_t)(c * NCHUNK) * D + ks * KSLICE;
      #pragma unroll
      for (int it = 0; it < (NCHUNK * KSLICE / 8) / 256; ++it) {  // 4 iters of 16B x2
        int f  = (tid + it * 256) * 8;           // bf16 element within 128x64 tile
        int r  = f >> 6, kk = f & (KSLICE - 1);
        size_t go = (size_t)r * D + kk;
        int    lo = r * SB + kk;
        cp16_g2l(gh + go, sBhi + lo);
        cp16_g2l(gl + go, sBlo + lo);
      }
      wait_async();
      __syncthreads();

      #pragma unroll
      for (int k2 = 0; k2 < 2; ++k2) {               // two K=32 steps per slice
        const int ka = ks * KSLICE + k2 * 32 + half * 16;  // A frag base K (global)
        const int kb = k2 * 32 + half * 16;                // B frag base K (slice-local)
        // A fragments (16 contiguous K per lane, matching B: elems 0..15 @ka)
        v16bf ah0 = ld_frag(sAhi + (am0      ) * SA + ka, 8);
        v16bf al0 = ld_frag(sAlo + (am0      ) * SA + ka, 8);
        v16bf ah1 = ld_frag(sAhi + (am0 + 16 ) * SA + ka, 8);
        v16bf al1 = ld_frag(sAlo + (am0 + 16 ) * SA + ka, 8);
        #pragma unroll
        for (int nt = 0; nt < 4; ++nt) {
          const __bf16* bp = sBhi + (bnb + nt * 16) * SB + kb;
          const __bf16* lp = sBlo + (bnb + nt * 16) * SB + kb;
          v16bf bh = ld_frag(bp, 8);   // B frag: 16 contiguous K per lane, same order as A
          v16bf bl = ld_frag(lp, 8);
          // bf16x3 emulated f32 product
          acc[nt]     = wmma_bf16(ah0, bh, acc[nt]);
          acc[nt]     = wmma_bf16(ah0, bl, acc[nt]);
          acc[nt]     = wmma_bf16(al0, bh, acc[nt]);
          acc[4 + nt] = wmma_bf16(ah1, bh, acc[4 + nt]);
          acc[4 + nt] = wmma_bf16(ah1, bl, acc[4 + nt]);
          acc[4 + nt] = wmma_bf16(al1, bh, acc[4 + nt]);
        }
      }
    }

    // ---- distance + running argmin (n ascending -> strict '<' keeps first index) ----
    #pragma unroll
    for (int nt = 0; nt < 4; ++nt) {
      int n = c * NCHUNK + wn * 64 + nt * 16 + l16;
      float e2 = enorm[n];
      #pragma unroll
      for (int mt = 0; mt < 2; ++mt) {
        #pragma unroll
        for (int j = 0; j < 8; ++j) {
          float dv = e2 - 2.0f * acc[mt * 4 + nt][j];
          int bi = mt * 8 + j;
          if (dv < bestv[bi]) { bestv[bi] = dv; besti[bi] = n; }
        }
      }
    }
  }

  // ---- cross-lane reduction within each 16-lane half (rows fixed per half) ----
  __syncthreads();   // done reading B region; safe to reuse as sRed
  #pragma unroll
  for (int mt = 0; mt < 2; ++mt) {
    #pragma unroll
    for (int j = 0; j < 8; ++j) {
      float v = bestv[mt * 8 + j];
      int  ix = besti[mt * 8 + j];
      #pragma unroll
      for (int off = 8; off >= 1; off >>= 1) {
        float ov = __shfl_xor(v, off, 16);
        int   oi = __shfl_xor(ix, off, 16);
        if (ov < v || (ov == v && oi < ix)) { v = ov; ix = oi; }
      }
      if (l16 == 0) {
        int rloc = wm * 32 + mt * 16 + half * 8 + j;
        sRed[rloc * 2 + wn] = make_float2(v, (float)ix);
      }
    }
  }
  __syncthreads();

  // ---- combine the two N-half waves, emit indices ----
  if (tid < MT) {
    float2 a = sRed[tid * 2 + 0];
    float2 b = sRed[tid * 2 + 1];
    int ia = (int)a.y, ib = (int)b.y;
    bool pb = (b.x < a.x) || (b.x == a.x && ib < ia);
    int code = pb ? ib : ia;
    int grow = blk * MT + tid;
    idx_out[grow]  = code;
    idxf_out[grow] = (float)code;
  }
}

// ---------------- kernel 2: gather z_q = codebook[idx] (exact f32) ----------------
__global__ __launch_bounds__(256) void vq_gather_kernel(const float* __restrict__ cb,
                                                        const int* __restrict__ idx,
                                                        float* __restrict__ out) {
  size_t g = (size_t)blockIdx.x * 256 + threadIdx.x;
  size_t e = g * 4;                    // element index into (16384 x 512)
  int row = (int)(e >> 9);
  int k   = (int)(e & (D - 1));
  int code = idx[row];
  float4 v = *(const float4*)(cb + (size_t)code * D + k);
  *(float4*)(out + e) = v;
}

// ---------------- host launcher ----------------
extern "C" void kernel_launch(void* const* d_in, const int* in_sizes, int n_in,
                              void* d_out, int out_size, void* d_ws, size_t ws_size,
                              hipStream_t stream) {
  (void)in_sizes; (void)n_in; (void)out_size; (void)ws_size;
  const float* z  = (const float*)d_in[0];   // (8,2048,512) f32
  const float* cb = (const float*)d_in[1];   // (8192,512)  f32
  float* out = (float*)d_out;                // [8388608 z_q][16384 indices-as-float]

  // workspace: [cbhi 8MB][cblo 8MB][enorm 32KB][idx 64KB]
  __bf16* cbhi  = (__bf16*)d_ws;
  __bf16* cblo  = cbhi + (size_t)NE * D;
  float*  enorm = (float*)(cblo + (size_t)NE * D);
  int*    idxbuf = (int*)(enorm + NE);

  // one-time codebook pre-conversion to planar bf16 hi/lo + exact norms
  vq_cvt_kernel<<<(NE * D / 4) / 256, 256, 0, stream>>>(cb, cbhi, cblo);
  vq_enorm_kernel<<<NE / 256, 256, 0, stream>>>(cb, enorm);

  // fused distance GEMM + argmin; dynamic LDS: A hi/lo + B hi/lo
  size_t shmem = ((size_t)2 * MT * SA + (size_t)2 * NCHUNK * SB) * 2; // bytes (bf16)
  vq_argmin_kernel<<<MROWS / MT, 256, shmem, stream>>>(z, cbhi, cblo, enorm, idxbuf,
                                                       out + (size_t)MROWS * D);

  // z_q gather (straight-through forward value == codebook rows exactly)
  vq_gather_kernel<<<(MROWS * D / 4) / 256, 256, 0, stream>>>(cb, idxbuf, out);
}
